// ContextualLoss_40673340293859
// MI455X (gfx1250) — matrix operand order repaired and checked by reading, not compiled, not run on hardware
//
#include <hip/hip_runtime.h>
#include <hip/hip_bf16.h>

typedef _Float16 h16v __attribute__((ext_vector_type(16)));
typedef _Float16 h8v  __attribute__((ext_vector_type(8)));
typedef float    f8v  __attribute__((ext_vector_type(8)));

#define HW    4096
#define CH    256
#define NB    4
#define EPSV  1e-5f
#define CT    32            // columns per LDS B tile
#define NT    (HW / CT)     // 128 tiles per sweep
#define BPITCH 264          // padded halfs per column in LDS (conflict-free b128 reads)

#if __has_builtin(__builtin_amdgcn_sched_barrier)
#define SCHED_FENCE() __builtin_amdgcn_sched_barrier(0)
#else
#define SCHED_FENCE()
#endif

// ---------------- channel mean of y over (n, h, w) ----------------
__global__ void k_channel_mean(const float* __restrict__ y, float* __restrict__ mu) {
    __shared__ float red[256];
    const int c = blockIdx.x;
    const int t = threadIdx.x;
    float s = 0.f;
    for (int l = t; l < NB * HW; l += 256) {
        int n  = l >> 12;
        int hw = l & (HW - 1);
        s += y[((size_t)n * CH + c) * HW + hw];
    }
    red[t] = s; __syncthreads();
    for (int o = 128; o > 0; o >>= 1) { if (t < o) red[t] += red[t + o]; __syncthreads(); }
    if (t == 0) mu[c] = red[0] * (1.0f / (NB * HW));
}

// --- center by y_mu, L2-normalize over C, transpose to fp16 [n][hw][c] ---
__global__ void __launch_bounds__(256)
k_normalize(const float* __restrict__ x, const float* __restrict__ y,
            const float* __restrict__ mu,
            _Float16* __restrict__ xn, _Float16* __restrict__ yn) {
    __shared__ _Float16 T[64][258];          // pitch 258 halfs = 129 dw -> bank = c' (conflict-free)
    const int t   = threadIdx.x;
    const int n   = blockIdx.y;
    const int hw0 = blockIdx.x * 256;
    const float* xb = x + (size_t)n * CH * HW + hw0;
    const float* yb = y + (size_t)n * CH * HW + hw0;

    float ssx = 0.f, ssy = 0.f;
    for (int c = 0; c < CH; ++c) {
        float m  = mu[c];
        float xv = xb[(size_t)c * HW + t] - m;
        float yv = yb[(size_t)c * HW + t] - m;
        ssx += xv * xv; ssy += yv * yv;
    }
    const float xs = 1.0f / fmaxf(sqrtf(ssx), 1e-12f);
    const float ys = 1.0f / fmaxf(sqrtf(ssy), 1e-12f);

    const int cp = t & 63;
    const int hb = t >> 6;
    for (int cc = 0; cc < CH; cc += 64) {
        for (int c2 = 0; c2 < 64; ++c2) {
            int c = cc + c2;
            T[c2][t] = (_Float16)((xb[(size_t)c * HW + t] - mu[c]) * xs);
        }
        __syncthreads();
        for (int g = 0; g < 64; ++g) {
            int hwp = g * 4 + hb;
            xn[((size_t)n * HW + hw0 + hwp) * CH + cc + cp] = T[cp][hwp];
        }
        __syncthreads();
        for (int c2 = 0; c2 < 64; ++c2) {
            int c = cc + c2;
            T[c2][t] = (_Float16)((yb[(size_t)c * HW + t] - mu[c]) * ys);
        }
        __syncthreads();
        for (int g = 0; g < 64; ++g) {
            int hwp = g * 4 + hb;
            yn[((size_t)n * HW + hw0 + hwp) * CH + cc + cp] = T[cp][hwp];
        }
        __syncthreads();
    }
}

__global__ void k_zero(float* __restrict__ p, int n) {
    int i = blockIdx.x * blockDim.x + threadIdx.x;
    if (i < n) p[i] = 0.f;
}

// ---------------- cross-lane reductions within 16-lane groups ----------------
__device__ __forceinline__ float red16_max(float v) {
    v = fmaxf(v, __shfl_xor(v, 1, 32));
    v = fmaxf(v, __shfl_xor(v, 2, 32));
    v = fmaxf(v, __shfl_xor(v, 4, 32));
    v = fmaxf(v, __shfl_xor(v, 8, 32));
    return v;
}
__device__ __forceinline__ float red16_sum(float v) {
    v += __shfl_xor(v, 1, 32);
    v += __shfl_xor(v, 2, 32);
    v += __shfl_xor(v, 4, 32);
    v += __shfl_xor(v, 8, 32);
    return v;
}

// ---- fused triple-pass GEMM: S = xn·ynᵀ per batch; rowmax → rowsum → colmax(cx) ----
__global__ void __launch_bounds__(256)
k_cx_gemm(const _Float16* __restrict__ xn, const _Float16* __restrict__ yn,
          float* __restrict__ colmax) {
    __shared__ __align__(16) _Float16 Bs[2][CT * BPITCH];

    const int t     = threadIdx.x;
    const int lane  = t & 31;
    const int wave  = t >> 5;
    const int hl    = lane >> 4;
    const int l16   = lane & 15;
    const int strip = blockIdx.x;     // 0..31 (128-row strips)
    const int batch = blockIdx.y;     // 0..3

    const _Float16* Ab = xn + ((size_t)batch * HW + strip * 128 + wave * 16 + l16) * CH;
    const _Float16* Bg = yn + (size_t)batch * HW * CH;

    // A panel (16 rows x 256 K) resident in registers for all three passes
    h16v afrag[8];
    #pragma unroll
    for (int kb = 0; kb < 8; ++kb) {
        afrag[kb] = *(const h16v*)(Ab + kb * 32 + hl * 16);
    }

    // async global->LDS copy of one 32-col x 256-K tile (16KB), 4 chunks/thread
    auto issue_tile = [&](int tile, int buf) {
        const _Float16* src = Bg + (size_t)tile * CT * CH;
        #pragma unroll
        for (int i = 0; i < 4; ++i) {
            int ch  = i * 256 + t;
            int col = ch >> 5;
            int off = ch & 31;
            unsigned la = (unsigned)(unsigned long long)&Bs[buf][col * BPITCH + off * 8];
            unsigned long long ga =
                (unsigned long long)(const void*)(src + (size_t)col * CH + off * 8);
            asm volatile("global_load_async_to_lds_b128 %0, %1, off"
                         :: "v"(la), "v"(ga) : "memory");
        }
    };

    float pmax[8], psum[8], inv2[8], invr[8];
    #pragma unroll
    for (int v = 0; v < 8; ++v) { pmax[v] = -2.0f; psum[v] = 0.f; inv2[v] = 0.f; invr[v] = 0.f; }

    for (int pass = 0; pass < 3; ++pass) {
        issue_tile(0, 0);
        for (int tile = 0; tile < NT; ++tile) {
            const int cur = tile & 1;
            if (tile + 1 < NT) {
                issue_tile(tile + 1, cur ^ 1);
                asm volatile("s_wait_asynccnt 0x4" ::: "memory");
            } else {
                asm volatile("s_wait_asynccnt 0x0" ::: "memory");
            }
            __syncthreads();

            #pragma unroll
            for (int st = 0; st < CT / 16; ++st) {
                const _Float16* bp = &Bs[cur][(st * 16 + l16) * BPITCH + hl * 16];
                // preload the whole subtile's B fragments; the sched_barrier pins
                // all 16 ds_load_b128 ahead of the WMMA chain so there is a single
                // s_wait_dscnt and 8 back-to-back WMMAs (D->C accumulation has no
                // A/B RAW hazard)
                h16v bf[8];
                #pragma unroll
                for (int kb = 0; kb < 8; ++kb) bf[kb] = *(const h16v*)(bp + kb * 32);
                SCHED_FENCE();
                f8v d = {};
                #pragma unroll
                for (int kb = 0; kb < 8; ++kb)
                    d = __builtin_amdgcn_wmma_f32_16x16x32_f16(false, afrag[kb], false, bf[kb],
                                                               (short)0, d, false, false);
                SCHED_FENCE();
                if (pass == 0) {
                    #pragma unroll
                    for (int v = 0; v < 8; ++v) pmax[v] = fmaxf(pmax[v], d[v]);
                } else if (pass == 1) {
                    #pragma unroll
                    for (int v = 0; v < 8; ++v)
                        psum[v] += __expf(2.0f - (1.0f - d[v]) * inv2[v]);
                } else {
                    float cm = 0.f;
                    #pragma unroll
                    for (int v = 0; v < 8; ++v)
                        cm = fmaxf(cm, __expf(2.0f - (1.0f - d[v]) * inv2[v]) * invr[v]);
                    cm = fmaxf(cm, __shfl_xor(cm, 16, 32));
                    if (hl == 0) {
                        int j = tile * CT + st * 16 + l16;
                        atomicMax((unsigned int*)&colmax[batch * HW + j],
                                  __float_as_uint(cm));       // cx > 0: uint cmp == float cmp
                    }
                }
            }
            __syncthreads();   // protect buffer reuse before next issue
        }
        if (pass == 0) {
            #pragma unroll
            for (int v = 0; v < 8; ++v)
                inv2[v] = 2.0f / ((1.0f - red16_max(pmax[v])) + EPSV);
        } else if (pass == 1) {
            #pragma unroll
            for (int v = 0; v < 8; ++v) invr[v] = 1.0f / red16_sum(psum[v]);
        }
    }
}

// ---------------- scalar epilogue: mean over j, -log, mean over batch ----------------
__global__ void k_final(const float* __restrict__ colmax, float* __restrict__ out) {
    __shared__ float red[256];
    const int t = threadIdx.x;
    float acc = 0.f;
    for (int n = 0; n < NB; ++n) {
        float s = 0.f;
        for (int j = t; j < HW; j += 256) s += colmax[n * HW + j];
        red[t] = s; __syncthreads();
        for (int o = 128; o > 0; o >>= 1) { if (t < o) red[t] += red[t + o]; __syncthreads(); }
        if (t == 0) acc += -__logf(red[0] * (1.0f / HW) + EPSV);
        __syncthreads();
    }
    if (t == 0) out[0] = acc * (1.0f / NB);
}

extern "C" void kernel_launch(void* const* d_in, const int* in_sizes, int n_in,
                              void* d_out, int out_size, void* d_ws, size_t ws_size,
                              hipStream_t stream) {
    const float* x = (const float*)d_in[0];
    const float* y = (const float*)d_in[1];
    float* out = (float*)d_out;

    // workspace: mu[256] | colmax[NB*HW] | xn fp16 | yn fp16   (~16.9 MB)
    float* mu     = (float*)d_ws;
    float* colmax = mu + 256;
    _Float16* xn  = (_Float16*)(colmax + NB * HW);
    _Float16* yn  = xn + (size_t)NB * HW * CH;

    k_channel_mean<<<CH, 256, 0, stream>>>(y, mu);
    dim3 gn(HW / 256, NB);
    k_normalize<<<gn, 256, 0, stream>>>(x, y, mu, xn, yn);
    k_zero<<<(NB * HW + 255) / 256, 256, 0, stream>>>(colmax, NB * HW);
    dim3 g(HW / 128, NB);
    k_cx_gemm<<<g, 256, 0, stream>>>(xn, yn, colmax);
    k_final<<<1, 256, 0, stream>>>(colmax, out);
}
